// ChromaticTransportEvaluator_53730040873694
// MI455X (gfx1250) — matrix operand order repaired and checked by reading, not compiled, not run on hardware
//
#include <hip/hip_runtime.h>
#include <math.h>

// ---------------------------------------------------------------------------
// ChromaticTransportEvaluator for MI455X (gfx1250, wave32, WMMA + TDM)
// Heavy phase: Sinkhorn matvecs vs shared symmetric 512x512 K matrix,
// bf16 WMMA 16x16x32 (f32 accum). Per-block A-tile staged into LDS by the
// Tensor Data Mover (tensor_load_to_lds, TENSORcnt), removing the 8x
// redundant A loads across the block's waves. B streams from L2.
// ---------------------------------------------------------------------------

typedef __attribute__((ext_vector_type(16))) __bf16 v16bf;
typedef __attribute__((ext_vector_type(8)))  __bf16 v8bf;
typedef __attribute__((ext_vector_type(8)))  float  v8f;
typedef __attribute__((ext_vector_type(4)))  unsigned int v4u;
typedef __attribute__((ext_vector_type(4)))  int    v4i;
typedef __attribute__((ext_vector_type(8)))  int    v8i;

#define NGRID   512          // 8^3 bins
#define NPATCH  1024         // 4 * 16 * 16
#define REG_F   0.1f
#define EPS_F   1e-6f
#define ITERS   20

// ---- workspace layout (bytes) ----
static const size_t HR_OFF  = 0;                        // 1024*512 f32 = 2 MiB
static const size_t HT_OFF  = HR_OFF  + (size_t)NPATCH*NGRID*4;
static const size_t UF_OFF  = HT_OFF  + (size_t)NPATCH*NGRID*4;   // u  f32
static const size_t VF_OFF  = UF_OFF  + (size_t)NPATCH*NGRID*4;   // v  f32
static const size_t UB_OFF  = VF_OFF  + (size_t)NPATCH*NGRID*4;   // u  bf16
static const size_t VB_OFF  = UB_OFF  + (size_t)NPATCH*NGRID*2;   // v  bf16
static const size_t KB_OFF  = VB_OFF  + (size_t)NPATCH*NGRID*2;   // K  bf16 512x512
static const size_t KC_OFF  = KB_OFF  + (size_t)NGRID*NGRID*2;    // K*cost bf16
static const size_t EMD_OFF = KC_OFF  + (size_t)NGRID*NGRID*2;    // emd f32 (1024)
static const size_t MM_OFF  = EMD_OFF + 4096;                     // 16 u32 min/max
static const size_t CAV_OFF = MM_OFF  + 1024;                     // 24 f32 centers

// monotonic float <-> uint encoding for atomic min/max
__device__ __forceinline__ unsigned fenc(float x) {
  unsigned u = __float_as_uint(x);
  return (u & 0x80000000u) ? ~u : (u | 0x80000000u);
}
__device__ __forceinline__ float fdec(unsigned e) {
  unsigned u = (e & 0x80000000u) ? (e ^ 0x80000000u) : ~e;
  return __uint_as_float(u);
}

__device__ __forceinline__ void srgb2oklab(float r, float g, float b, float lab[3]) {
  float x[3] = {r, g, b}, lin[3];
#pragma unroll
  for (int c = 0; c < 3; ++c) {
    float v = fminf(fmaxf(x[c], 0.f), 1.f);
    lin[c] = (v <= 0.04045f) ? v * (1.0f / 12.92f)
                             : powf((v + 0.055f) * (1.0f / 1.055f), 2.4f);
  }
  float L = fmaxf(0.4122214708f*lin[0] + 0.5363325363f*lin[1] + 0.0514459929f*lin[2], 0.f);
  float M = fmaxf(0.2119034982f*lin[0] + 0.6806995451f*lin[1] + 0.1073969566f*lin[2], 0.f);
  float S = fmaxf(0.0883024619f*lin[0] + 0.2817188376f*lin[1] + 0.6299787005f*lin[2], 0.f);
  float l = cbrtf(fmaxf(L, 1e-12f));
  float m = cbrtf(fmaxf(M, 1e-12f));
  float s = cbrtf(fmaxf(S, 1e-12f));
  lab[0] = 0.2104542553f*l + 0.793617785f *m - 0.0040720468f*s;
  lab[1] = 1.9779984951f*l - 2.428592205f *m + 0.4505937099f*s;
  lab[2] = 0.0259040371f*l + 0.7827717662f*m - 0.808675766f *s;
}

// ---------------------------------------------------------------------------
// TDM: 2D tile load (16 rows x 512 bf16, row stride 512) global -> LDS.
// D# built per CDNA5 ISA 8.3/8.4: group0 {count=1, lds_addr, global_addr,
// type=2}, group1 {data_size=1(2B), tensor_dim0=512, tensor_dim1=1024,
// tile_dim0=512, tile_dim1=16, tensor_dim0_stride=512}. Groups 2/3 zero (2D).
// ---------------------------------------------------------------------------
__device__ __forceinline__ void tdm_load_tile16(const __bf16* gsrc, void* lds_dst) {
  unsigned long long ga = (unsigned long long)(uintptr_t)gsrc;
  unsigned lds = (unsigned)(uintptr_t)lds_dst;     // low 32 bits = LDS offset
  v4u g0;
  g0.x = 1u;                                        // count=1, user descriptor
  g0.y = lds;                                       // [63:32]  lds_addr
  g0.z = (unsigned)(ga & 0xFFFFFFFFu);              // [95:64]  global_addr lo
  g0.w = (unsigned)((ga >> 32) & 0x01FFFFFFu)       // [120:96] global_addr hi
       | (2u << 30);                                // [127:126] type=2 (image)
  v8i g1;
  g1[0] = (int)(1u << 16);                          // data_size=1 -> 2 bytes
  g1[1] = (int)((512u & 0xFFFFu) << 16);            // tensor_dim0[15:0] @ [63:48]
  g1[2] = (int)((512u >> 16) | ((1024u & 0xFFFFu) << 16)); // td0 hi | td1 lo
  g1[3] = (int)((1024u >> 16) | (512u << 16));      // td1 hi | tile_dim0=512
  g1[4] = (int)16u;                                 // tile_dim1=16, tile_dim2=0
  g1[5] = (int)512;                                 // tensor_dim0_stride lo32
  g1[6] = 0;                                        // stride hi | td1_stride lo
  g1[7] = 0;                                        // td1_stride hi
  v4i gz = {0, 0, 0, 0};
#if __clang_major__ >= 23
  v8i gz8 = {0, 0, 0, 0, 0, 0, 0, 0};
  __builtin_amdgcn_tensor_load_to_lds(g0, g1, gz, gz, gz8, 0);
#else
  __builtin_amdgcn_tensor_load_to_lds(g0, g1, gz, gz, 0);
#endif
}

// ---------------------------------------------------------------------------
// init: zero hists/emd, u = 1, min/max sentinels.  grid: 2048 x 256
// ---------------------------------------------------------------------------
__global__ void k_init(float* hr, float* ht, float* uf, __bf16* ub,
                       float* emd, unsigned* mm) {
  size_t i = (size_t)blockIdx.x * 256 + threadIdx.x;
  hr[i] = 0.f; ht[i] = 0.f;
  uf[i] = 1.0f; ub[i] = (__bf16)1.0f;
  if (i < NPATCH) emd[i] = 0.f;
  if (i < 16)     mm[i] = (i & 1) ? 0u : 0xFFFFFFFFu;
}

// ---------------------------------------------------------------------------
// per-channel OKLab min/max per image (LDS tree + encoded atomics)
// grid: 2048 x 256
// ---------------------------------------------------------------------------
__global__ void k_minmax(const float* __restrict__ ref, const float* __restrict__ tgt,
                         unsigned* __restrict__ mm) {
  __shared__ float smn[3][256];
  __shared__ float smx[3][256];
  int tid = threadIdx.x;
  size_t idx = (size_t)blockIdx.x * 256 + tid;
  int img = (int)(idx >> 18);
  int rem = (int)(idx & 262143);
  int b = rem >> 16;
  int pix = rem & 65535;
  int y = pix >> 8, x = pix & 255;
  const float* im = img ? tgt : ref;
  const float* p = im + (size_t)b * 3 * 65536 + y * 256 + x;
  float lab[3];
  srgb2oklab(p[0], p[65536], p[131072], lab);
#pragma unroll
  for (int c = 0; c < 3; ++c) { smn[c][tid] = lab[c]; smx[c][tid] = lab[c]; }
  __syncthreads();
  for (int s = 128; s > 0; s >>= 1) {
    if (tid < s) {
#pragma unroll
      for (int c = 0; c < 3; ++c) {
        smn[c][tid] = fminf(smn[c][tid], smn[c][tid + s]);
        smx[c][tid] = fmaxf(smx[c][tid], smx[c][tid + s]);
      }
    }
    __syncthreads();
  }
  if (tid == 0) {
    unsigned* m = mm + img * 8;
#pragma unroll
    for (int c = 0; c < 3; ++c) {
      atomicMin(&m[c * 2 + 0], fenc(smn[c][0]));
      atomicMax(&m[c * 2 + 1], fenc(smx[c][0]));
    }
  }
}

// ---------------------------------------------------------------------------
// averaged separable grid centers: cav[c*8+k]
// ---------------------------------------------------------------------------
__global__ void k_centers(const unsigned* __restrict__ mm, float* __restrict__ cav) {
  int t = threadIdx.x;
  if (t >= 24) return;
  int c = t / 8, k = t % 8;
  float acc = 0.f;
#pragma unroll
  for (int img = 0; img < 2; ++img) {
    float lo = fdec(mm[img * 8 + c * 2 + 0]) - 0.01f;
    float hi = fdec(mm[img * 8 + c * 2 + 1]) + 0.01f;
    if (hi - lo < 1e-4f) { lo -= 0.05f; hi += 0.05f; }
    acc += lo + (hi - lo) * ((k + 0.5f) / 8.0f);
  }
  cav[c * 8 + k] = 0.5f * acc;
}

// ---------------------------------------------------------------------------
// histograms: nearest point of a separable uniform-midpoint grid == binning
// grid: 2048 x 256
// ---------------------------------------------------------------------------
__global__ void k_hist(const float* __restrict__ ref, const float* __restrict__ tgt,
                       const unsigned* __restrict__ mm,
                       float* __restrict__ hr, float* __restrict__ ht) {
  size_t idx = (size_t)blockIdx.x * 256 + threadIdx.x;
  int img = (int)(idx >> 18);
  int rem = (int)(idx & 262143);
  int b = rem >> 16;
  int pix = rem & 65535;
  int y = pix >> 8, x = pix & 255;
  const float* im = img ? tgt : ref;
  const float* p = im + (size_t)b * 3 * 65536 + y * 256 + x;
  float lab[3];
  srgb2oklab(p[0], p[65536], p[131072], lab);
  int bin[3];
#pragma unroll
  for (int c = 0; c < 3; ++c) {
    float lo = fdec(mm[img * 8 + c * 2 + 0]) - 0.01f;
    float hi = fdec(mm[img * 8 + c * 2 + 1]) + 0.01f;
    if (hi - lo < 1e-4f) { lo -= 0.05f; hi += 0.05f; }
    int bc = (int)((lab[c] - lo) * (8.0f / (hi - lo)));
    bin[c] = min(7, max(0, bc));
  }
  int patch = b * 256 + (y >> 4) * 16 + (x >> 4);
  int n = bin[0] * 64 + bin[1] * 8 + bin[2];
  float* h = img ? ht : hr;
  atomicAdd(&h[(size_t)patch * NGRID + n], 1.0f / 256.0f);
}

// ---------------------------------------------------------------------------
// K = exp(-cost/REG), KC = K*cost (512x512 symmetric, bf16).  grid: 1024x256
// ---------------------------------------------------------------------------
__global__ void k_cost(const float* __restrict__ cav,
                       __bf16* __restrict__ Kb, __bf16* __restrict__ KCb) {
  int idx = blockIdx.x * 256 + threadIdx.x;
  int i = idx >> 9, j = idx & 511;
  int iv[3] = { i >> 6, (i >> 3) & 7, i & 7 };
  int jv[3] = { j >> 6, (j >> 3) & 7, j & 7 };
  float d2 = 0.f;
#pragma unroll
  for (int c = 0; c < 3; ++c) {
    float d = cav[c * 8 + iv[c]] - cav[c * 8 + jv[c]];
    d2 += d * d;
  }
  float cost = (d2 > 0.f) ? sqrtf(d2) : 0.f;
  float K = expf(-cost * (1.0f / REG_F));
  Kb[idx]  = (__bf16)K;
  KCb[idx] = (__bf16)(K * cost);
}

// ---------------------------------------------------------------------------
// One Sinkhorn half-step: out = h / (in @ K + eps)   (K symmetric)
// Block: 8 waves, tile M=16 x N=128. A-tile (16x512 bf16 = 16 KB) staged to
// LDS once per block via TDM; B streams from global (L2-resident).
// grid: (4, 64) x 256 threads
// ---------------------------------------------------------------------------
__global__ void __launch_bounds__(256)
k_sink_step(const __bf16* __restrict__ inb, const float* __restrict__ h,
            float* __restrict__ outf, __bf16* __restrict__ outb,
            const __bf16* __restrict__ Kmat) {
  __shared__ __bf16 As[16 * NGRID];
  int lane  = threadIdx.x & 31;
  int w     = threadIdx.x >> 5;
  int jbase = (blockIdx.x * 8 + w) * 16;
  int mbase = blockIdx.y * 16;
  int l16 = lane & 15, lh = lane >> 4;

  if (threadIdx.x < 32) {
    tdm_load_tile16(inb + (size_t)mbase * NGRID, As);
    __builtin_amdgcn_s_wait_tensorcnt(0);
  }
  __syncthreads();

  const __bf16* arow = As   + l16 * NGRID + lh * 8;
  const __bf16* brow = Kmat + (size_t)(jbase + l16) * NGRID + lh * 8;

  v8f c = {0.f, 0.f, 0.f, 0.f, 0.f, 0.f, 0.f, 0.f};
#pragma unroll
  for (int kk = 0; kk < NGRID; kk += 32) {
    v8bf a0 = *(const v8bf*)(arow + kk);
    v8bf a1 = *(const v8bf*)(arow + kk + 16);
    v8bf b0 = *(const v8bf*)(brow + kk);
    v8bf b1 = *(const v8bf*)(brow + kk + 16);
    v16bf A = __builtin_shufflevector(a0, a1, 0,1,2,3,4,5,6,7,8,9,10,11,12,13,14,15);
    v16bf B = __builtin_shufflevector(b0, b1, 0,1,2,3,4,5,6,7,8,9,10,11,12,13,14,15);
    c = __builtin_amdgcn_wmma_f32_16x16x32_bf16(false, A, false, B, (short)0, c,
                                                false, false);
  }

  int col = jbase + l16;
#pragma unroll
  for (int r = 0; r < 8; ++r) {
    int row = mbase + r + (lh << 3);
    float val = h[(size_t)row * NGRID + col] / (c[r] + EPS_F);
    outf[(size_t)row * NGRID + col] = val;
    outb[(size_t)row * NGRID + col] = (__bf16)val;
  }
}

// ---------------------------------------------------------------------------
// emd[p] = sum_j (u @ KC)[p,j] * v[p,j] : same WMMA tile, atomic row-reduce
// ---------------------------------------------------------------------------
__global__ void __launch_bounds__(256)
k_emd(const __bf16* __restrict__ ub, const float* __restrict__ vf,
      const __bf16* __restrict__ KCmat, float* __restrict__ emd) {
  __shared__ __bf16 As[16 * NGRID];
  int lane  = threadIdx.x & 31;
  int w     = threadIdx.x >> 5;
  int jbase = (blockIdx.x * 8 + w) * 16;
  int mbase = blockIdx.y * 16;
  int l16 = lane & 15, lh = lane >> 4;

  if (threadIdx.x < 32) {
    tdm_load_tile16(ub + (size_t)mbase * NGRID, As);
    __builtin_amdgcn_s_wait_tensorcnt(0);
  }
  __syncthreads();

  const __bf16* arow = As    + l16 * NGRID + lh * 8;
  const __bf16* brow = KCmat + (size_t)(jbase + l16) * NGRID + lh * 8;

  v8f c = {0.f, 0.f, 0.f, 0.f, 0.f, 0.f, 0.f, 0.f};
#pragma unroll
  for (int kk = 0; kk < NGRID; kk += 32) {
    v8bf a0 = *(const v8bf*)(arow + kk);
    v8bf a1 = *(const v8bf*)(arow + kk + 16);
    v8bf b0 = *(const v8bf*)(brow + kk);
    v8bf b1 = *(const v8bf*)(brow + kk + 16);
    v16bf A = __builtin_shufflevector(a0, a1, 0,1,2,3,4,5,6,7,8,9,10,11,12,13,14,15);
    v16bf B = __builtin_shufflevector(b0, b1, 0,1,2,3,4,5,6,7,8,9,10,11,12,13,14,15);
    c = __builtin_amdgcn_wmma_f32_16x16x32_bf16(false, A, false, B, (short)0, c,
                                                false, false);
  }

  int col = jbase + l16;
#pragma unroll
  for (int r = 0; r < 8; ++r) {
    int row = mbase + r + (lh << 3);
    float val = c[r] * vf[(size_t)row * NGRID + col];
    atomicAdd(&emd[row], val);
  }
}

// ---------------------------------------------------------------------------
// half-pixel bilinear upsample (16x16 -> 256x256) with nan_to_num guard
// grid: 1024 x 256
// ---------------------------------------------------------------------------
__global__ void k_resize(const float* __restrict__ emd, float* __restrict__ out) {
  int idx = blockIdx.x * 256 + threadIdx.x;
  int b = idx >> 16;
  int pix = idx & 65535;
  int y = pix >> 8, x = pix & 255;
  float sy = (y + 0.5f) * (1.0f / 16.0f) - 0.5f;
  float sx = (x + 0.5f) * (1.0f / 16.0f) - 0.5f;
  int y0 = (int)floorf(sy), x0 = (int)floorf(sx);
  float fy = sy - (float)y0, fx = sx - (float)x0;
  int y0c = min(15, max(0, y0)),     x0c = min(15, max(0, x0));
  int y1c = min(15, max(0, y0 + 1)), x1c = min(15, max(0, x0 + 1));
  const float* e = emd + b * 256;
  float v00 = e[y0c * 16 + x0c]; v00 = isfinite(v00) ? v00 : 0.f;
  float v01 = e[y0c * 16 + x1c]; v01 = isfinite(v01) ? v01 : 0.f;
  float v10 = e[y1c * 16 + x0c]; v10 = isfinite(v10) ? v10 : 0.f;
  float v11 = e[y1c * 16 + x1c]; v11 = isfinite(v11) ? v11 : 0.f;
  out[idx] = (1.f - fy) * ((1.f - fx) * v00 + fx * v01)
           +        fy  * ((1.f - fx) * v10 + fx * v11);
}

// ---------------------------------------------------------------------------
extern "C" void kernel_launch(void* const* d_in, const int* in_sizes, int n_in,
                              void* d_out, int out_size, void* d_ws, size_t ws_size,
                              hipStream_t stream) {
  (void)in_sizes; (void)n_in; (void)out_size; (void)ws_size;
  const float* ref = (const float*)d_in[0];
  const float* tgt = (const float*)d_in[1];
  char* ws = (char*)d_ws;

  float*    hr  = (float*)   (ws + HR_OFF);
  float*    ht  = (float*)   (ws + HT_OFF);
  float*    uf  = (float*)   (ws + UF_OFF);
  float*    vf  = (float*)   (ws + VF_OFF);
  __bf16*   ub  = (__bf16*)  (ws + UB_OFF);
  __bf16*   vb  = (__bf16*)  (ws + VB_OFF);
  __bf16*   Kb  = (__bf16*)  (ws + KB_OFF);
  __bf16*   KCb = (__bf16*)  (ws + KC_OFF);
  float*    emd = (float*)   (ws + EMD_OFF);
  unsigned* mm  = (unsigned*)(ws + MM_OFF);
  float*    cav = (float*)   (ws + CAV_OFF);
  float*    out = (float*)d_out;

  k_init   <<<2048, 256, 0, stream>>>(hr, ht, uf, ub, emd, mm);
  k_minmax <<<2048, 256, 0, stream>>>(ref, tgt, mm);
  k_centers<<<1,    64,  0, stream>>>(mm, cav);
  k_hist   <<<2048, 256, 0, stream>>>(ref, tgt, mm, hr, ht);
  k_cost   <<<1024, 256, 0, stream>>>(cav, Kb, KCb);

  dim3 sgrid(4, 64);
  for (int it = 0; it < ITERS; ++it) {
    k_sink_step<<<sgrid, 256, 0, stream>>>(ub, ht, vf, vb, Kb);   // v = ht/(K u + eps)
    k_sink_step<<<sgrid, 256, 0, stream>>>(vb, hr, uf, ub, Kb);   // u = hr/(K v + eps)
  }
  k_emd    <<<sgrid, 256, 0, stream>>>(ub, vf, KCb, emd);
  k_resize <<<1024, 256, 0, stream>>>(emd, out);
}
